// TransitionModel_43138651521707
// MI455X (gfx1250) — matrix operand order repaired and checked by reading, not compiled, not run on hardware
//
#include <hip/hip_runtime.h>
#include <hip/hip_bf16.h>
#include <stdint.h>
#include <stddef.h>

// ---------------- types ----------------
typedef __bf16 bf16;
typedef __attribute__((ext_vector_type(16))) __bf16 v16bf;
typedef __attribute__((ext_vector_type(8)))  __bf16 v8bf;
typedef __attribute__((ext_vector_type(4)))  __bf16 v4bf;
typedef __attribute__((ext_vector_type(8)))  float  v8f;

// ---------------- problem dims ----------------
#define T_STEPS 50
#define BATCH   256
#define BT      16          // batch rows per workgroup (one WMMA M tile)
#define NW      16          // waves per workgroup
#define NTHREADS (NW * 32)  // 512
#define LDA     1032        // padded LDS row stride (bf16 elems) for 1024-wide buffers

// ---------------- bf16 weight workspace layout (elements) ----------------
#define OFF_WSA   0u          // 1024 x 160
#define OFF_WIH   163840u     // 3072 x 1024
#define OFF_WHH   3309568u    // 3072 x 1024
#define OFF_WPRH  6455296u    // 1024 x 1024
#define OFF_WPRS  7503872u    //  256 x 1024
#define OFF_WPOH  7766016u    // 1024 x 2048
#define OFF_WPOS  9863168u    //  256 x 1024

// ---------------- output layout (float elements) ----------------
static constexpr size_t SZ_BEL = (size_t)T_STEPS * BATCH * 1024;
static constexpr size_t SZ_SM  = (size_t)T_STEPS * BATCH * 128;
static constexpr size_t OUT_BEL    = 0;
static constexpr size_t OUT_PRI_S  = SZ_BEL;
static constexpr size_t OUT_PRI_M  = SZ_BEL + 1 * SZ_SM;
static constexpr size_t OUT_PRI_SD = SZ_BEL + 2 * SZ_SM;
static constexpr size_t OUT_POS_S  = SZ_BEL + 3 * SZ_SM;
static constexpr size_t OUT_POS_M  = SZ_BEL + 4 * SZ_SM;
static constexpr size_t OUT_POS_SD = SZ_BEL + 5 * SZ_SM;

// ---------------- scalar helpers (fast-path HW transcendentals) ----------------
__device__ __forceinline__ bf16 f2bf(float f)  { return (bf16)f; }
__device__ __forceinline__ float bf2f(bf16 b)  { return (float)b; }
__device__ __forceinline__ float frcp(float x) { return __builtin_amdgcn_rcpf(x); }
__device__ __forceinline__ float sigm(float x) { return frcp(1.0f + __expf(-x)); }
__device__ __forceinline__ float elu1(float x) { return x > 0.f ? x : __expf(x) - 1.f; }
__device__ __forceinline__ float splus(float x){ return fmaxf(x, 0.f) + __logf(1.f + __expf(-fabsf(x))); }
__device__ __forceinline__ float tanh_f(float x){
  float e = __expf(2.f * x);
  return 1.f - 2.f * frcp(e + 1.f);
}

// ---------------- WMMA helpers ----------------
__device__ __forceinline__ v8f wmma_bf16(v16bf a, v16bf b, v8f c) {
  return __builtin_amdgcn_wmma_f32_16x16x32_bf16(false, a, false, b, (short)0, c, false, false);
}

// A fragment: 16x32 bf16 tile, row-major activations in LDS (row stride lda).
__device__ __forceinline__ v16bf lda_frag(const bf16* A, int lda, int kb, int lane) {
  const int row = lane & 15;
  const int hi  = (lane >> 4) & 1;
  const bf16* p = A + row * lda + kb + hi * 8;
  union { v16bf v; v8bf h[2]; } u;
  u.h[0] = *(const v8bf*)(p);
  u.h[1] = *(const v8bf*)(p + 16);
  return u.v;
}
// B fragment: B[k,n] = W[n,k] for row-major W[N,K] in global memory.
__device__ __forceinline__ v16bf ldb_frag(const bf16* Wt, int ldw, int kb, int lane) {
  const int n  = lane & 15;
  const int hi = (lane >> 4) & 1;
  const bf16* p = Wt + (size_t)n * ldw + kb + hi * 8;
  union { v16bf v; v8bf h[2]; } u;
  u.h[0] = *(const v8bf*)(p);
  u.h[1] = *(const v8bf*)(p + 16);
  return u.v;
}
__device__ __forceinline__ v8f gemm16(const bf16* Alds, int lda,
                                      const bf16* Wt, int ldw,
                                      int ktiles, int lane) {
  v8f c = {};
  for (int kt = 0; kt < ktiles; ++kt) {
    v16bf a = lda_frag(Alds, lda, kt * 32, lane);
    v16bf b = ldb_frag(Wt, ldw, kt * 32, lane);
    c = wmma_bf16(a, b, c);
  }
  return c;
}

// ---------------- prologue: fp32 -> bf16 weight pack ----------------
__global__ void convert_f32_bf16(const float* __restrict__ src, bf16* __restrict__ dst, int n) {
  int i = blockIdx.x * blockDim.x + threadIdx.x;
  if (i < n) dst[i] = (bf16)src[i];
}

// ---------------- persistent RSSM scan kernel ----------------
__global__ __launch_bounds__(NTHREADS)
void rssm_persistent(const float* __restrict__ prev_state,    // [256,128]
                     const float* __restrict__ actions,       // [50,256,32]
                     const float* __restrict__ prev_belief,   // [256,1024]
                     const float* __restrict__ observations,  // [50,256,1024]
                     const float* __restrict__ nonterminals,  // [50,256,1]
                     const float* __restrict__ prior_noise,   // [50,256,128]
                     const float* __restrict__ posterior_noise,
                     const bf16*  __restrict__ wsb,           // packed bf16 weights
                     const float* __restrict__ b_sa,  const float* __restrict__ b_ih,
                     const float* __restrict__ b_hh,  const float* __restrict__ b_prh,
                     const float* __restrict__ b_prs, const float* __restrict__ b_poh,
                     const float* __restrict__ b_pos,
                     float* __restrict__ out) {
  __shared__ bf16 Hld[16 * LDA];   // belief/h carry (bf16)
  __shared__ bf16 OBS[16 * LDA];   // current observation tile
  __shared__ bf16 XHP[16 * LDA];   // x, later hp (x dead after gate GEMM)
  __shared__ bf16 HQL[16 * LDA];   // h staging, later hq
  __shared__ bf16 SA [16 * 160];   // [state*nt, action]
  __shared__ bf16 SC [16 * 128];   // posterior state carry

  const int tid  = threadIdx.x;
  const int lane = tid & 31;
  const int wave = tid >> 5;
  const int bb   = blockIdx.x * BT;

  // ---- init carries (vectorized fp32x4 -> bf16x4) ----
  for (int i = tid; i < 16 * 256; i += NTHREADS) {
    int r = i >> 8, c4 = (i & 255) * 4;
    float4 v = *(const float4*)(prev_belief + (size_t)(bb + r) * 1024 + c4);
    v4bf o = { f2bf(v.x), f2bf(v.y), f2bf(v.z), f2bf(v.w) };
    *(v4bf*)(&Hld[r * LDA + c4]) = o;
  }
  for (int i = tid; i < 16 * 32; i += NTHREADS) {
    int r = i >> 5, c4 = (i & 31) * 4;
    float4 v = *(const float4*)(prev_state + (size_t)(bb + r) * 128 + c4);
    v4bf o = { f2bf(v.x), f2bf(v.y), f2bf(v.z), f2bf(v.w) };
    *(v4bf*)(&SC[r * 128 + c4]) = o;
  }
  __syncthreads();

  for (int t = 0; t < T_STEPS; ++t) {
    // Opaque OFFSET (not pointer!) keeps kernarg/global address-space
    // provenance (=> global_load, SGPR base) while still preventing the
    // compiler from hoisting weight-fragment loads out of the t loop.
    size_t toff = 0;
    asm volatile("" : "+s"(toff));
    const bf16* wsl = wsb + toff;
    const bf16* Wsa  = wsl + OFF_WSA;
    const bf16* Wih  = wsl + OFF_WIH;
    const bf16* Whh  = wsl + OFF_WHH;
    const bf16* Wprh = wsl + OFF_WPRH;
    const bf16* Wprs = wsl + OFF_WPRS;
    const bf16* Wpoh = wsl + OFF_WPOH;
    const bf16* Wpos = wsl + OFF_WPOS;

    // ---- stage sa = [state*nt, action] and obs (fp32 -> bf16) ----
    for (int i = tid; i < 16 * 160; i += NTHREADS) {
      int r = i / 160, c = i - r * 160;
      float nt = nonterminals[(size_t)t * BATCH + bb + r];
      float v  = (c < 128) ? bf2f(SC[r * 128 + c]) * nt
                           : actions[((size_t)t * BATCH + bb + r) * 32 + (c - 128)];
      SA[i] = f2bf(v);
    }
    for (int i = tid; i < 16 * 256; i += NTHREADS) {
      int r = i >> 8, c4 = (i & 255) * 4;
      float4 v = *(const float4*)(observations + ((size_t)t * BATCH + bb + r) * 1024 + c4);
      v4bf o = { f2bf(v.x), f2bf(v.y), f2bf(v.z), f2bf(v.w) };
      *(v4bf*)(&OBS[r * LDA + c4]) = o;
    }
    __syncthreads();

    // ---- GEMM1: x = elu(sa @ Wsa.T + b_sa), K=160 (5 ktiles) ----
    for (int ti = 0; ti < 4; ++ti) {
      const int ntile = wave * 4 + ti;
      v8f c = gemm16(SA, 160, Wsa + (size_t)ntile * 16 * 160, 160, 5, lane);
      const int ncol = ntile * 16 + (lane & 15);
      const int mb   = (lane >> 4) * 8;
      const float bias = b_sa[ncol];
#pragma unroll
      for (int r = 0; r < 8; ++r)
        XHP[(mb + r) * LDA + ncol] = f2bf(elu1(c[r] + bias));
    }
    __syncthreads();

    // ---- GEMM2: fused GRU gates -> h (per-wave: 4 N tiles x 6 accumulators) ----
    for (int ti = 0; ti < 4; ++ti) {
      const int ntile = wave * 4 + ti;
      v8f gir = {}, giz = {}, gin = {}, ghr = {}, ghz = {}, ghn = {};
      const bf16* Wir = Wih + (size_t)(0    + ntile * 16) * 1024;
      const bf16* Wiz = Wih + (size_t)(1024 + ntile * 16) * 1024;
      const bf16* Win = Wih + (size_t)(2048 + ntile * 16) * 1024;
      const bf16* Whr = Whh + (size_t)(0    + ntile * 16) * 1024;
      const bf16* Whz = Whh + (size_t)(1024 + ntile * 16) * 1024;
      const bf16* Whn = Whh + (size_t)(2048 + ntile * 16) * 1024;
      for (int kt = 0; kt < 32; ++kt) {
        const int kb = kt * 32;
        v16bf ax = lda_frag(XHP, LDA, kb, lane);
        v16bf ah = lda_frag(Hld, LDA, kb, lane);
        gir = wmma_bf16(ax, ldb_frag(Wir, 1024, kb, lane), gir);
        giz = wmma_bf16(ax, ldb_frag(Wiz, 1024, kb, lane), giz);
        gin = wmma_bf16(ax, ldb_frag(Win, 1024, kb, lane), gin);
        ghr = wmma_bf16(ah, ldb_frag(Whr, 1024, kb, lane), ghr);
        ghz = wmma_bf16(ah, ldb_frag(Whz, 1024, kb, lane), ghz);
        ghn = wmma_bf16(ah, ldb_frag(Whn, 1024, kb, lane), ghn);
      }
      const int ncol = ntile * 16 + (lane & 15);
      const int mb   = (lane >> 4) * 8;
      const float bir = b_ih[ncol], biz = b_ih[1024 + ncol], bin = b_ih[2048 + ncol];
      const float bhr = b_hh[ncol], bhz = b_hh[1024 + ncol], bhn = b_hh[2048 + ncol];
#pragma unroll
      for (int r = 0; r < 8; ++r) {
        const int m = mb + r;
        float rg = sigm(gir[r] + bir + ghr[r] + bhr);
        float zg = sigm(giz[r] + biz + ghz[r] + bhz);
        float ng = tanh_f(gin[r] + bin + rg * (ghn[r] + bhn));
        float hp = bf2f(Hld[m * LDA + ncol]);
        float h  = (1.f - zg) * ng + zg * hp;
        HQL[m * LDA + ncol] = f2bf(h);   // stage h through idle HQL
        out[OUT_BEL + ((size_t)t * BATCH + bb + m) * 1024 + ncol] = h;
      }
    }
    __syncthreads();   // everyone done reading old belief; h staged in HQL
    for (int i = tid; i < 16 * 256; i += NTHREADS) {
      int r = i >> 8, c4 = (i & 255) * 4;
      *(v4bf*)(&Hld[r * LDA + c4]) = *(const v4bf*)(&HQL[r * LDA + c4]);
    }
    __syncthreads();   // h carry committed

    // ---- GEMM3: hp = elu(h @ Wprh.T + b_prh) -> XHP (x is dead) ----
    for (int ti = 0; ti < 4; ++ti) {
      const int ntile = wave * 4 + ti;
      v8f c = gemm16(Hld, LDA, Wprh + (size_t)ntile * 16 * 1024, 1024, 32, lane);
      const int ncol = ntile * 16 + (lane & 15);
      const int mb   = (lane >> 4) * 8;
      const float bias = b_prh[ncol];
#pragma unroll
      for (int r = 0; r < 8; ++r)
        XHP[(mb + r) * LDA + ncol] = f2bf(elu1(c[r] + bias));
    }
    __syncthreads();

    // ---- GEMM5: hq = elu([h,obs] @ Wpoh.T + b_poh), K=2048 over two LDS bufs ----
    for (int ti = 0; ti < 4; ++ti) {
      const int ntile = wave * 4 + ti;
      const bf16* Wt = Wpoh + (size_t)ntile * 16 * 2048;
      v8f c = {};
      for (int kt = 0; kt < 64; ++kt) {
        const int kb = kt * 32;
        v16bf a = (kt < 32) ? lda_frag(Hld, LDA, kb, lane)
                            : lda_frag(OBS, LDA, kb - 1024, lane);
        c = wmma_bf16(a, ldb_frag(Wt, 2048, kb, lane), c);
      }
      const int ncol = ntile * 16 + (lane & 15);
      const int mb   = (lane >> 4) * 8;
      const float bias = b_poh[ncol];
#pragma unroll
      for (int r = 0; r < 8; ++r)
        HQL[(mb + r) * LDA + ncol] = f2bf(elu1(c[r] + bias));
    }
    __syncthreads();

    // ---- heads: waves 0-7 -> prior, waves 8-15 -> posterior (wave-uniform) ----
    if (wave < 8) {
      const int w = wave;
      v8f cm = gemm16(XHP, LDA, Wprs + (size_t)(w * 16) * 1024, 1024, 32, lane);
      v8f cs = gemm16(XHP, LDA, Wprs + (size_t)(128 + w * 16) * 1024, 1024, 32, lane);
      const int col = w * 16 + (lane & 15);
      const int mb  = (lane >> 4) * 8;
      const float bm = b_prs[col], bs = b_prs[128 + col];
#pragma unroll
      for (int r = 0; r < 8; ++r) {
        const int m = mb + r;
        const size_t idx = ((size_t)t * BATCH + bb + m) * 128 + col;
        float mp = cm[r] + bm;
        float sp = splus(cs[r] + bs) + 0.1f;
        out[OUT_PRI_M + idx]  = mp;
        out[OUT_PRI_SD + idx] = sp;
        out[OUT_PRI_S + idx]  = mp + sp * prior_noise[idx];
      }
    } else {
      const int w = wave - 8;
      v8f cm = gemm16(HQL, LDA, Wpos + (size_t)(w * 16) * 1024, 1024, 32, lane);
      v8f cs = gemm16(HQL, LDA, Wpos + (size_t)(128 + w * 16) * 1024, 1024, 32, lane);
      const int col = w * 16 + (lane & 15);
      const int mb  = (lane >> 4) * 8;
      const float bm = b_pos[col], bs = b_pos[128 + col];
#pragma unroll
      for (int r = 0; r < 8; ++r) {
        const int m = mb + r;
        const size_t idx = ((size_t)t * BATCH + bb + m) * 128 + col;
        float mq = cm[r] + bm;
        float sq = splus(cs[r] + bs) + 0.1f;
        float ps = mq + sq * posterior_noise[idx];
        out[OUT_POS_M + idx]  = mq;
        out[OUT_POS_SD + idx] = sq;
        out[OUT_POS_S + idx]  = ps;
        SC[m * 128 + col] = f2bf(ps);   // state carry for t+1
      }
    }
    __syncthreads();
  }
}

// ---------------- launch ----------------
extern "C" void kernel_launch(void* const* d_in, const int* in_sizes, int n_in,
                              void* d_out, int out_size, void* d_ws, size_t ws_size,
                              hipStream_t stream) {
  (void)in_sizes; (void)n_in; (void)out_size; (void)ws_size;
  const float* prev_state      = (const float*)d_in[0];
  const float* actions         = (const float*)d_in[1];
  const float* prev_belief     = (const float*)d_in[2];
  const float* observations    = (const float*)d_in[3];
  const float* nonterminals    = (const float*)d_in[4];
  const float* prior_noise     = (const float*)d_in[5];
  const float* posterior_noise = (const float*)d_in[6];
  const float* W_sa  = (const float*)d_in[7];
  const float* b_sa  = (const float*)d_in[8];
  const float* W_ih  = (const float*)d_in[9];
  const float* b_ih  = (const float*)d_in[10];
  const float* W_hh  = (const float*)d_in[11];
  const float* b_hh  = (const float*)d_in[12];
  const float* W_prh = (const float*)d_in[13];
  const float* b_prh = (const float*)d_in[14];
  const float* W_prs = (const float*)d_in[15];
  const float* b_prs = (const float*)d_in[16];
  const float* W_poh = (const float*)d_in[17];
  const float* b_poh = (const float*)d_in[18];
  const float* W_pos = (const float*)d_in[19];
  const float* b_pos = (const float*)d_in[20];

  bf16* W = (bf16*)d_ws;
  auto cv = [&](const float* src, unsigned off, int n) {
    convert_f32_bf16<<<(n + 255) / 256, 256, 0, stream>>>(src, W + off, n);
  };
  cv(W_sa,  OFF_WSA,  1024 * 160);
  cv(W_ih,  OFF_WIH,  3072 * 1024);
  cv(W_hh,  OFF_WHH,  3072 * 1024);
  cv(W_prh, OFF_WPRH, 1024 * 1024);
  cv(W_prs, OFF_WPRS, 256 * 1024);
  cv(W_poh, OFF_WPOH, 1024 * 2048);
  cv(W_pos, OFF_WPOS, 256 * 1024);

  rssm_persistent<<<BATCH / BT, NTHREADS, 0, stream>>>(
      prev_state, actions, prev_belief, observations, nonterminals,
      prior_noise, posterior_noise, W,
      b_sa, b_ih, b_hh, b_prh, b_prs, b_poh, b_pos,
      (float*)d_out);
}
